// GeneralMACE_5162550690017
// MI455X (gfx1250) — matrix operand order repaired and checked by reading, not compiled, not run on hardware
//
#include <hip/hip_runtime.h>
#include <hip/hip_fp16.h>
#include <math.h>

// ---------------- problem constants (match reference) ----------------
#define FCH   64          // channels F
#define HD    4           // hidden irreps dim (l=0,1)
#define IDIM  16          // interaction irreps dim (l=0..3)
#define NPE   10          // edge TP paths
#define NPB   23          // node TP paths (inter->inter)
#define NPH   10          // node TP paths (inter->hid)
#define NSPEC 10
#define RMAXF 5.0f
#define EPSF  0.5f
#define RWCOL (NPE * FCH) // 640

typedef __attribute__((ext_vector_type(16))) _Float16 v16h;
typedef __attribute__((ext_vector_type(8)))  _Float16 v8h;
typedef __attribute__((ext_vector_type(8)))  float    v8f;
typedef unsigned int u32;
typedef __attribute__((ext_vector_type(4))) u32 u32x4;
typedef __attribute__((ext_vector_type(8))) int i32x8;
typedef __attribute__((ext_vector_type(4))) int i32x4;

#if defined(__has_builtin)
#  if __has_builtin(__builtin_amdgcn_tensor_load_to_lds) && __has_builtin(__builtin_amdgcn_s_wait_tensorcnt)
#    define USE_TDM 1
#  endif
#endif
#ifndef USE_TDM
#  define USE_TDM 0
#endif

struct PathT { int l1, l2, l3; };
static constexpr PathT PATHS_E[NPE] = {
  {0,0,0},{0,1,1},{0,2,2},{0,3,3},{1,0,1},{1,1,0},{1,1,2},{1,2,1},{1,2,3},{1,3,2}};
static constexpr PathT PATHS_B[NPB] = {
  {0,0,0},{0,1,1},{0,2,2},{0,3,3},
  {1,0,1},{1,1,0},{1,1,2},{1,2,1},{1,2,3},{1,3,2},
  {2,0,2},{2,1,1},{2,1,3},{2,2,0},{2,2,2},{2,3,1},{2,3,3},
  {3,0,3},{3,1,2},{3,2,1},{3,2,3},{3,3,0},{3,3,2}};
static constexpr PathT PATHS_H[NPH] = {
  {0,0,0},{0,1,1},{1,0,1},{1,1,0},{1,2,1},{2,1,1},{2,2,0},{2,3,1},{3,2,1},{3,3,0}};
static constexpr int LOFF[4] = {0,1,4,9};

// real-basis Clebsch-Gordan surrogate (selection-rule sparse; constant-folds)
__device__ __forceinline__ float cg_val(int l1, int l2, int l3, int a, int b, int c) {
  const int m1 = a - l1, m2 = b - l2, m3 = c - l3;
  const int am1 = m1 < 0 ? -m1 : m1, am2 = m2 < 0 ? -m2 : m2, am3 = m3 < 0 ? -m3 : m3;
  const int dd = am1 > am2 ? am1 - am2 : am2 - am1;
  if (am3 != am1 + am2 && am3 != dd) return 0.0f;
  const float sgn = ((a + b + c) & 1) ? -1.0f : 1.0f;
  return sgn / sqrtf((float)((2 * l1 + 1) * (2 * l2 + 1)));
}

// ---------------- Tensor Data Mover: 2D f16 tile -> LDS ----------------
// D# per ISA ch.8: group0 {count=1, lds_addr, global_addr, type=2},
// group1 {data_size=2B, tensor_dim0/1, tile_dim0/1, tensor_dim0_stride}.
// clang-23 toolchain: 6-arg builtin (g0, g1, g2, g3, g4, cpol).
__device__ __forceinline__ void tdm_load_2d_f16(u32 lds_off, const void* gptr,
                                                u32 tdim0, u32 tdim1,
                                                u32 tile0, u32 tile1, u32 stride0) {
#if USE_TDM
  unsigned long long ga = (unsigned long long)gptr;
  u32x4 g0;
  g0[0] = 0x1u;                                    // count=1 (valid user descriptor)
  g0[1] = lds_off;                                 // LDS byte address
  g0[2] = (u32)(ga & 0xFFFFFFFFu);                 // global_addr[31:0]
  g0[3] = (u32)((ga >> 32) & 0x01FFFFFFu) | (2u << 30);  // addr[56:32], type=2
  i32x8 g1;
  g1[0] = (int)(1u << 16);                         // data_size=1 -> 2 bytes
  g1[1] = (int)((tdim0 & 0xFFFFu) << 16);          // tensor_dim0[15:0] @ bits 63:48
  g1[2] = (int)(((tdim0 >> 16) & 0xFFFFu) | ((tdim1 & 0xFFFFu) << 16));
  g1[3] = (int)(((tdim1 >> 16) & 0xFFFFu) | (tile0 << 16));   // tile_dim0
  g1[4] = (int)(tile1 & 0xFFFFu);                  // tile_dim1 (tile_dim2=0)
  g1[5] = (int)stride0;                            // tensor_dim0_stride[31:0]
  g1[6] = 0;
  g1[7] = 0;
  i32x4 z4 = {0, 0, 0, 0};
  i32x8 z8 = {0, 0, 0, 0, 0, 0, 0, 0};
  __builtin_amdgcn_tensor_load_to_lds(g0, g1, z4, z4, z8, 0);
#else
  (void)lds_off; (void)gptr; (void)tdim0; (void)tdim1; (void)tile0; (void)tile1; (void)stride0;
#endif
}

// ---------------- vectorized WMMA fragment loads from LDS ----------------
// A fragment (16x32 f16 layout): lane row m, halves at k-base 8*(lane/16):
//   two contiguous 8xf16 runs -> 2x ds_load_b128
__device__ __forceinline__ v16h frag_a(const _Float16* As, int m, int k0, int half) {
  const v8h lo = *(const v8h*)&As[m * 64 + k0 + half * 8];
  const v8h hi = *(const v8h*)&As[m * 64 + k0 + 16 + half * 8];
  return __builtin_shufflevector(lo, hi, 0,1,2,3,4,5,6,7,8,9,10,11,12,13,14,15);
}
// B fragment from transposed LDS BsT[col][k] (col-major tile):
//   two contiguous 8xf16 runs at k-base 8*(lane/16) and 16+8*(lane/16) -> 2x ds_load_b128
__device__ __forceinline__ v16h frag_b(const _Float16* BsT, int col, int k0, int half) {
  const v8h lo = *(const v8h*)&BsT[col * 64 + k0 + half * 8];
  const v8h hi = *(const v8h*)&BsT[col * 64 + k0 + 16 + half * 8];
  return __builtin_shufflevector(lo, hi, 0,1,2,3,4,5,6,7,8,9,10,11,12,13,14,15);
}

// =====================================================================
// Generic per-l channel-mixing GEMM (weights shared across rows):
//   C[r, f] = sum_g A(r,g) * W[f,g],  r=(n,c): n=r/ADIV, c=r%ADIV
// One wave per 16x32 C tile, full K=64 staged, 4 static v_wmma.
// =====================================================================
template <int ADIV>
__global__ __launch_bounds__(32)
void wmma_lin(const float* __restrict__ A, long anst, long akst, long aoff,
              const float* __restrict__ Bw,
              float* __restrict__ C, long cnst, long cfst, long coff) {
  __shared__ __align__(16) _Float16 As[16 * 64];   // [m][k]
  __shared__ __align__(16) _Float16 BsT[32 * 64];  // [col][k]  (transposed tile)
  const int tm = blockIdx.x * 16, tn = blockIdx.y * 32;
  const int lane = threadIdx.x, m = lane & 15, half = lane >> 4;
  // A: lane covers row (tm+m), k-range half*32.. (one div per lane), b128 LDS stores
  {
    const int r = tm + m;
    const int nn = r / ADIV, cc = r % ADIV;
    const size_t abase = (size_t)nn * anst + cc + aoff;
    #pragma unroll
    for (int c8 = 0; c8 < 4; ++c8) {
      v8h tmp;
      #pragma unroll
      for (int e = 0; e < 8; ++e)
        tmp[e] = (_Float16)A[abase + (size_t)(half * 32 + c8 * 8 + e) * akst];
      *(v8h*)&As[m * 64 + half * 32 + c8 * 8] = tmp;
    }
  }
  // B: lane = column; weights contiguous in k -> straight row copy into BsT
  {
    const float* src = Bw + (size_t)(tn + lane) * 64;
    #pragma unroll
    for (int c8 = 0; c8 < 8; ++c8) {
      v8h tmp;
      #pragma unroll
      for (int e = 0; e < 8; ++e) tmp[e] = (_Float16)src[c8 * 8 + e];
      *(v8h*)&BsT[lane * 64 + c8 * 8] = tmp;
    }
  }
  __syncthreads();
  v8f acc0 = {}, acc1 = {};
  #pragma unroll
  for (int k0 = 0; k0 < 64; k0 += 32) {
    const v16h av = frag_a(As, m, k0, half);
    const v16h b0 = frag_b(BsT, m, k0, half);
    const v16h b1 = frag_b(BsT, 16 + m, k0, half);
    acc0 = __builtin_amdgcn_wmma_f32_16x16x32_f16(false, av, false, b0, (short)0, acc0, false, false);
    acc1 = __builtin_amdgcn_wmma_f32_16x16x32_f16(false, av, false, b1, (short)0, acc1, false, false);
  }
  #pragma unroll
  for (int v = 0; v < 8; ++v) {
    const int r = tm + half * 8 + v;
    const int nn = r / ADIV, cc = r % ADIV;
    const size_t cbase = (size_t)nn * cnst + cc + coff;
    C[cbase + (size_t)(tn + m) * cfst]      = acc0[v];
    C[cbase + (size_t)(tn + 16 + m) * cfst] = acc1[v];
  }
}

// =====================================================================
// Radial-MLP layer 2: rw[E,640](f16) = radh[E,64](f16) @ w2[64,640]
// Weights pre-transposed to w2ht[640,64] (f16) so both TDM tiles are
// plain row-major 2D DMAs and all LDS fragment reads are b128.
// =====================================================================
__global__ __launch_bounds__(32)
void wmma_gemm_rad2(const _Float16* __restrict__ A, const _Float16* __restrict__ BwT,
                    _Float16* __restrict__ Crw, int E) {
  __shared__ __align__(16) _Float16 As[16 * 64];   // [m][k]
  __shared__ __align__(16) _Float16 BsT[32 * 64];  // [col][k]
  const int tm = blockIdx.x * 16, tn = blockIdx.y * 32;
  const int lane = threadIdx.x, m = lane & 15, half = lane >> 4;
#if USE_TDM
  // A tile: 16 rows x 64 k, row-major stride 64 (f16)
  tdm_load_2d_f16((u32)(uintptr_t)As, A + (size_t)tm * 64,
                  64u, (u32)(E - tm), 64u, 16u, 64u);
  // B tile (transposed weights): 32 rows (cols of W) x 64 k, stride 64
  tdm_load_2d_f16((u32)(uintptr_t)BsT, BwT + (size_t)tn * 64,
                  64u, (u32)(RWCOL - tn), 64u, 32u, 64u);
  __builtin_amdgcn_s_wait_tensorcnt(0);
#else
  {
    const v8h* srcA = (const v8h*)(A + (size_t)(tm + m) * 64);
    v8h* dstA = (v8h*)&As[m * 64];
    #pragma unroll
    for (int c8 = 0; c8 < 4; ++c8) dstA[half * 4 + c8] = srcA[half * 4 + c8];
    const v8h* srcB = (const v8h*)(BwT + (size_t)(tn + lane) * 64);
    v8h* dstB = (v8h*)&BsT[lane * 64];
    #pragma unroll
    for (int c8 = 0; c8 < 8; ++c8) dstB[c8] = srcB[c8];
  }
  __syncthreads();
#endif
  v8f acc0 = {}, acc1 = {};
  #pragma unroll
  for (int k0 = 0; k0 < 64; k0 += 32) {
    const v16h av = frag_a(As, m, k0, half);
    const v16h b0 = frag_b(BsT, m, k0, half);
    const v16h b1 = frag_b(BsT, 16 + m, k0, half);
    acc0 = __builtin_amdgcn_wmma_f32_16x16x32_f16(false, av, false, b0, (short)0, acc0, false, false);
    acc1 = __builtin_amdgcn_wmma_f32_16x16x32_f16(false, av, false, b1, (short)0, acc1, false, false);
  }
  #pragma unroll
  for (int v = 0; v < 8; ++v) {
    const size_t r = (size_t)(tm + half * 8 + v);
    Crw[r * RWCOL + tn + m]      = (_Float16)acc0[v];
    Crw[r * RWCOL + tn + 16 + m] = (_Float16)acc1[v];
  }
}

// ---------------- geometry: r, spherical harmonics (l<=3), bessel ----------------
__global__ void k_edge_geom(const float* __restrict__ vec, float* __restrict__ sh,
                            float* __restrict__ rad, int E) {
  int e = blockIdx.x * blockDim.x + threadIdx.x;
  if (e >= E) return;
  float vx = vec[e * 3 + 0], vy = vec[e * 3 + 1], vz = vec[e * 3 + 2];
  float r = sqrtf(vx * vx + vy * vy + vz * vz + 1e-12f);
  float inv = 1.0f / r;
  float x = vx * inv, y = vy * inv, z = vz * inv;
  size_t b = (size_t)e * 16;
  sh[b + 0] = 1.0f;
  const float s3 = 1.7320508f;
  sh[b + 1] = s3 * y; sh[b + 2] = s3 * z; sh[b + 3] = s3 * x;
  const float s15 = 3.8729833f;
  float xx = x * x, yy = y * y, zz = z * z;
  sh[b + 4] = s15 * x * y;
  sh[b + 5] = s15 * y * z;
  sh[b + 6] = 1.1180340f * (3.0f * zz - 1.0f);
  sh[b + 7] = s15 * x * z;
  sh[b + 8] = 0.5f * s15 * (xx - yy);
  sh[b + 9]  = 1.0458250f * y * (3.0f * xx - yy);
  sh[b + 10] = 5.1234753f * x * y * z;
  sh[b + 11] = 0.8100926f * y * (5.0f * zz - 1.0f);
  sh[b + 12] = 0.6614378f * z * (5.0f * zz - 3.0f);
  sh[b + 13] = 0.8100926f * x * (5.0f * zz - 1.0f);
  sh[b + 14] = 2.5617376f * z * (xx - yy);
  sh[b + 15] = 1.0458250f * x * (xx - 3.0f * yy);
  float xr = r * (1.0f / RMAXF);
  float x2 = xr * xr, x6 = x2 * x2 * x2;
  float env = 1.0f - 28.0f * x6 + 48.0f * x6 * xr - 21.0f * x6 * x2;
  if (xr >= 1.0f) env = 0.0f;
  const float cb = 0.63245553f;  // sqrt(2/R_MAX)
  #pragma unroll
  for (int k = 0; k < 8; ++k) {
    float arg = (float)(k + 1) * 3.14159265f * xr;
    rad[(size_t)e * 8 + k] = cb * __sinf(arg) * inv * env;
  }
}

// ---------------- node embedding ----------------
__global__ void k_embed(const float* __restrict__ wemb, const int* __restrict__ sp,
                        float* __restrict__ x, int N) {
  int idx = blockIdx.x * blockDim.x + threadIdx.x;
  if (idx >= N * FCH) return;
  int n = idx >> 6, f = idx & 63;
  float v = wemb[sp[n] * FCH + f];
  float* p = x + (size_t)idx * HD;
  p[0] = v; p[1] = 0.0f; p[2] = 0.0f; p[3] = 0.0f;
}

// ---------------- radial MLP layer 1 + silu (f16 output) ----------------
__global__ void k_rad1(const float* __restrict__ rad, const float* __restrict__ w1,
                       _Float16* __restrict__ radh, int E) {
  int idx = blockIdx.x * blockDim.x + threadIdx.x;
  if (idx >= E * FCH) return;
  int e = idx >> 6, j = idx & 63;
  float s = 0.0f;
  #pragma unroll
  for (int k = 0; k < 8; ++k) s += rad[(size_t)e * 8 + k] * w1[k * 64 + j];
  radh[idx] = (_Float16)(s / (1.0f + __expf(-s)));  // silu
}

// ---------------- w_rad2 [64,640] f32 -> transposed f16 [640,64] ----------------
__global__ void k_cvt16t(const float* __restrict__ in, _Float16* __restrict__ outT, int total) {
  int i = blockIdx.x * blockDim.x + threadIdx.x;
  if (i >= total) return;
  int n = i >> 6, k = i & 63;             // outT[n][k] = in[k][n]
  outT[i] = (_Float16)in[(size_t)k * RWCOL + n];
}

// ---------------- utility kernels ----------------
__global__ void k_zero(float* __restrict__ p, int n) {
  int i = blockIdx.x * blockDim.x + threadIdx.x;
  if (i < n) p[i] = 0.0f;
}
__global__ void k_scale(const float* __restrict__ in, float* __restrict__ out, float s, int n) {
  int i = blockIdx.x * blockDim.x + threadIdx.x;
  if (i < n) out[i] = s * in[i];
}
__global__ void k_add(float* __restrict__ a, const float* __restrict__ b, int n) {
  int i = blockIdx.x * blockDim.x + threadIdx.x;
  if (i < n) a[i] += b[i];
}

// ---------------- fused edge tensor product + scatter-add ----------------
__global__ __launch_bounds__(64)
void k_edge_tp(const float* __restrict__ h, const float* __restrict__ sh,
               const _Float16* __restrict__ rw, const int* __restrict__ senders,
               const int* __restrict__ receivers, float* __restrict__ agg) {
  const int e = blockIdx.x, f = threadIdx.x;
  __shared__ float shs[16];
  __shared__ float hs[FCH * HD];
  const int sid = senders[e], rid = receivers[e];
  __builtin_prefetch(&rw[(size_t)e * RWCOL], 0, 3);   // global_prefetch_b8
  if (f < 16) shs[f] = sh[(size_t)e * 16 + f];
  for (int t = f; t < FCH * HD; t += 64) hs[t] = h[(size_t)sid * (FCH * HD) + t];
  __syncthreads();
  float rwv[NPE];
  #pragma unroll
  for (int p = 0; p < NPE; ++p) rwv[p] = (float)rw[(size_t)e * RWCOL + p * FCH + f];
  float outv[IDIM];
  #pragma unroll
  for (int c = 0; c < IDIM; ++c) outv[c] = 0.0f;
  #pragma unroll
  for (int p = 0; p < NPE; ++p) {
    const int l1 = PATHS_E[p].l1, l2 = PATHS_E[p].l2, l3 = PATHS_E[p].l3;
    const float w = rwv[p];
    #pragma unroll
    for (int a = 0; a < 3; ++a) {
      if (a >= 2 * l1 + 1) break;
      #pragma unroll
      for (int b = 0; b < 7; ++b) {
        if (b >= 2 * l2 + 1) break;
        #pragma unroll
        for (int c = 0; c < 7; ++c) {
          if (c >= 2 * l3 + 1) break;
          float cg = cg_val(l1, l2, l3, a, b, c);
          if (cg != 0.0f)
            outv[LOFF[l3] + c] += w * cg * hs[f * HD + LOFF[l1] + a] * shs[LOFF[l2] + b];
        }
      }
    }
  }
  #pragma unroll
  for (int c = 0; c < IDIM; ++c)
    __hip_atomic_fetch_add(&agg[(size_t)rid * (FCH * IDIM) + f * IDIM + c], outv[c],
                           __ATOMIC_RELAXED, __HIP_MEMORY_SCOPE_AGENT);
}

// ---------------- species-dependent per-l channel mixing ----------------
__global__ __launch_bounds__(64)
void k_slin(const float* __restrict__ in, const float* __restrict__ w,
            const int* __restrict__ sp, float* __restrict__ outp,
            int D, int NL, float scale) {
  const int n = blockIdx.x, f = threadIdx.x;
  extern __shared__ float xs[];
  for (int t = f; t < FCH * D; t += 64) xs[t] = in[(size_t)n * FCH * D + t];
  __syncthreads();
  const int s = sp[n];
  for (int c = 0; c < D; ++c) {
    int l = (D == IDIM) ? (c < 1 ? 0 : (c < 4 ? 1 : (c < 9 ? 2 : 3))) : (c < 1 ? 0 : 1);
    const float* wr = w + (((size_t)s * NL + l) * FCH + f) * FCH;
    float acc = 0.0f;
    for (int g = 0; g < FCH; ++g) acc += wr[g] * xs[g * D + c];
    outp[(size_t)n * FCH * D + f * D + c] = scale * acc;
  }
}

// ---------------- correlation-3 equivariant product basis ----------------
__global__ void k_product(const float* __restrict__ a, const int* __restrict__ sp,
                          const float* __restrict__ wp1, const float* __restrict__ wpB,
                          const float* __restrict__ wp3, float* __restrict__ prod, int N) {
  int idx = blockIdx.x * blockDim.x + threadIdx.x;
  if (idx >= N * FCH) return;
  const int n = idx >> 6, f = idx & 63, s = sp[n];
  float av[IDIM];
  #pragma unroll
  for (int c = 0; c < IDIM; ++c) av[c] = a[(size_t)idx * IDIM + c];
  float Bv[IDIM];
  #pragma unroll
  for (int c = 0; c < IDIM; ++c) Bv[c] = 0.0f;
  #pragma unroll
  for (int p = 0; p < NPB; ++p) {
    const int l1 = PATHS_B[p].l1, l2 = PATHS_B[p].l2, l3 = PATHS_B[p].l3;
    const float w = wpB[((size_t)s * NPB + p) * FCH + f];
    #pragma unroll
    for (int aa = 0; aa < 7; ++aa) {
      if (aa >= 2 * l1 + 1) break;
      #pragma unroll
      for (int bb = 0; bb < 7; ++bb) {
        if (bb >= 2 * l2 + 1) break;
        #pragma unroll
        for (int cc = 0; cc < 7; ++cc) {
          if (cc >= 2 * l3 + 1) break;
          float cg = cg_val(l1, l2, l3, aa, bb, cc);
          if (cg != 0.0f)
            Bv[LOFF[l3] + cc] += w * cg * av[LOFF[l1] + aa] * av[LOFF[l2] + bb];
        }
      }
    }
  }
  float t3v[HD] = {0.0f, 0.0f, 0.0f, 0.0f};
  #pragma unroll
  for (int p = 0; p < NPH; ++p) {
    const int l1 = PATHS_H[p].l1, l2 = PATHS_H[p].l2, l3 = PATHS_H[p].l3;
    const float w = wp3[((size_t)s * NPH + p) * FCH + f];
    #pragma unroll
    for (int aa = 0; aa < 7; ++aa) {
      if (aa >= 2 * l1 + 1) break;
      #pragma unroll
      for (int bb = 0; bb < 7; ++bb) {
        if (bb >= 2 * l2 + 1) break;
        #pragma unroll
        for (int cc = 0; cc < 3; ++cc) {
          if (cc >= 2 * l3 + 1) break;
          float cg = cg_val(l1, l2, l3, aa, bb, cc);
          if (cg != 0.0f)
            t3v[LOFF[l3] + cc] += w * cg * Bv[LOFF[l1] + aa] * av[LOFF[l2] + bb];
        }
      }
    }
  }
  const float w1a = wp1[((size_t)s * 2 + 0) * FCH + f];
  const float w1b = wp1[((size_t)s * 2 + 1) * FCH + f];
  float* pr = prod + (size_t)idx * HD;
  pr[0] = w1a * av[0] + Bv[0] + t3v[0];
  #pragma unroll
  for (int c = 1; c < 4; ++c) pr[c] = w1b * av[c] + Bv[c] + t3v[c];
}

// ---------------- readouts ----------------
__global__ __launch_bounds__(64)
void k_readout0(const float* __restrict__ x, const float* __restrict__ w,
                float* __restrict__ out) {
  const int n = blockIdx.x, f = threadIdx.x;
  __shared__ float red[64];
  red[f] = x[(size_t)n * (FCH * HD) + f * HD] * w[f];
  __syncthreads();
  for (int s = 32; s > 0; s >>= 1) { if (f < s) red[f] += red[f + s]; __syncthreads(); }
  if (f == 0) out[n * 2 + 0] = red[0];
}
__global__ __launch_bounds__(64)
void k_readout1(const float* __restrict__ x, const float* __restrict__ wa,
                const float* __restrict__ wb, float* __restrict__ out) {
  const int n = blockIdx.x, f = threadIdx.x;
  __shared__ float xs[64];
  __shared__ float hsil[16];
  xs[f] = x[(size_t)n * (FCH * HD) + f * HD];
  __syncthreads();
  if (f < 16) {
    float s = 0.0f;
    for (int g = 0; g < 64; ++g) s += xs[g] * wa[g * 16 + f];
    hsil[f] = s / (1.0f + __expf(-s));
  }
  __syncthreads();
  if (f == 0) {
    float s = 0.0f;
    for (int j = 0; j < 16; ++j) s += hsil[j] * wb[j];
    out[n * 2 + 1] = s;
  }
}

// ---------------- host-side launch helpers ----------------
static inline void launch_lin(hipStream_t st,
                              const float* A, long anst, long akst, long aoff, int adiv,
                              const float* Bw, float* C, long cnst, long cfst, long coff,
                              int M, int Ncols) {
  dim3 g(M / 16, Ncols / 32);
  switch (adiv) {
    case 1: wmma_lin<1><<<g, 32, 0, st>>>(A, anst, akst, aoff, Bw, C, cnst, cfst, coff); break;
    case 3: wmma_lin<3><<<g, 32, 0, st>>>(A, anst, akst, aoff, Bw, C, cnst, cfst, coff); break;
    case 5: wmma_lin<5><<<g, 32, 0, st>>>(A, anst, akst, aoff, Bw, C, cnst, cfst, coff); break;
    default: wmma_lin<7><<<g, 32, 0, st>>>(A, anst, akst, aoff, Bw, C, cnst, cfst, coff); break;
  }
}

extern "C" void kernel_launch(void* const* d_in, const int* in_sizes, int n_in,
                              void* d_out, int out_size, void* d_ws, size_t ws_size,
                              hipStream_t stream) {
  (void)n_in; (void)out_size; (void)ws_size;
  const float* vectors   = (const float*)d_in[0];
  const int*   specie    = (const int*)d_in[1];
  const int*   senders   = (const int*)d_in[2];
  const int*   receivers = (const int*)d_in[3];
  const float* w_embed   = (const float*)d_in[4];
  const float* w_up      = (const float*)d_in[5];
  const float* w_rad1    = (const float*)d_in[6];
  const float* w_rad2    = (const float*)d_in[7];
  const float* w_down    = (const float*)d_in[8];
  const float* w_skip0   = (const float*)d_in[9];
  const float* w_skip1   = (const float*)d_in[10];
  const float* wp1       = (const float*)d_in[11];
  const float* wpB       = (const float*)d_in[12];
  const float* wp3       = (const float*)d_in[13];
  const float* w_prod    = (const float*)d_in[14];
  const float* w_ro0     = (const float*)d_in[15];
  const float* w_ro1a    = (const float*)d_in[16];
  const float* w_ro1b    = (const float*)d_in[17];
  float* out = (float*)d_out;

  const int E = in_sizes[0] / 3;
  const int N = in_sizes[1];

  // workspace layout (256B-aligned byte offsets)
  char* wsb = (char*)d_ws;
  size_t off = 0;
  auto walloc = [&](size_t bytes) -> void* {
    void* p = wsb + off; off += (bytes + 255) & ~(size_t)255; return p;
  };
  float*    sh   = (float*)   walloc((size_t)E * 16 * 4);
  float*    rad  = (float*)   walloc((size_t)E * 8 * 4);
  _Float16* radh = (_Float16*)walloc((size_t)E * FCH * 2);
  _Float16* w2ht = (_Float16*)walloc((size_t)RWCOL * 64 * 2);   // transposed f16 weights
  _Float16* rw   = (_Float16*)walloc((size_t)E * RWCOL * 2);
  float*    x    = (float*)   walloc((size_t)N * FCH * HD * 4);
  float*    h    = (float*)   walloc((size_t)N * FCH * HD * 4);
  float*    agg  = (float*)   walloc((size_t)N * FCH * IDIM * 4);
  float*    m    = (float*)   walloc((size_t)N * FCH * IDIM * 4);
  float*    abuf = (float*)   walloc((size_t)N * FCH * IDIM * 4);
  float*    prod = (float*)   walloc((size_t)N * FCH * HD * 4);
  float*    sc   = (float*)   walloc((size_t)N * FCH * HD * 4);

  k_edge_geom<<<(E + 255) / 256, 256, 0, stream>>>(vectors, sh, rad, E);
  k_embed<<<(N * FCH + 255) / 256, 256, 0, stream>>>(w_embed, specie, x, N);

  for (int i = 0; i < 2; ++i) {
    // h = per-l lin_up(x)  (WMMA)
    launch_lin(stream, x, FCH * HD, HD, 0, 1, w_up + (size_t)(i * 2 + 0) * FCH * FCH,
               h, FCH * HD, HD, 0, N, FCH);
    launch_lin(stream, x, FCH * HD, HD, 1, 3, w_up + (size_t)(i * 2 + 1) * FCH * FCH,
               h, FCH * HD, HD, 1, N * 3, FCH);
    // radial MLP: layer1 VALU (f16 out), layer2 TDM + WMMA
    k_cvt16t<<<(RWCOL * 64 + 255) / 256, 256, 0, stream>>>(w_rad2 + (size_t)i * 64 * RWCOL, w2ht, RWCOL * 64);
    k_rad1<<<(E * FCH + 255) / 256, 256, 0, stream>>>(rad, w_rad1 + (size_t)i * 8 * 64, radh, E);
    {
      dim3 g(E / 16, RWCOL / 32);
      wmma_gemm_rad2<<<g, 32, 0, stream>>>(radh, w2ht, rw, E);
    }
    // messages -> agg
    k_zero<<<(N * FCH * IDIM + 255) / 256, 256, 0, stream>>>(agg, N * FCH * IDIM);
    k_edge_tp<<<E, 64, 0, stream>>>(h, sh, rw, senders, receivers, agg);
    // m = per-l lin_down(agg)  (WMMA)
    for (int l = 0; l < 4; ++l) {
      int d = 2 * l + 1;
      launch_lin(stream, agg, FCH * IDIM, IDIM, LOFF[l], d,
                 w_down + (size_t)(i * 4 + l) * FCH * FCH,
                 m, FCH * IDIM, IDIM, LOFF[l], N * d, FCH);
    }
    if (i == 0) {
      k_slin<<<N, 64, FCH * IDIM * 4, stream>>>(m, w_skip0, specie, abuf, IDIM, 4, EPSF);
    } else {
      k_scale<<<(N * FCH * IDIM + 255) / 256, 256, 0, stream>>>(m, abuf, EPSF, N * FCH * IDIM);
      k_slin<<<N, 64, FCH * HD * 4, stream>>>(x, w_skip1, specie, sc, HD, 2, 1.0f);
    }
    k_product<<<(N * FCH + 255) / 256, 256, 0, stream>>>(
        abuf, specie, wp1 + (size_t)i * NSPEC * 2 * FCH,
        wpB + (size_t)i * NSPEC * NPB * FCH, wp3 + (size_t)i * NSPEC * NPH * FCH, prod, N);
    // x = per-l lin_prod(prod)  (WMMA)
    launch_lin(stream, prod, FCH * HD, HD, 0, 1, w_prod + (size_t)(i * 2 + 0) * FCH * FCH,
               x, FCH * HD, HD, 0, N, FCH);
    launch_lin(stream, prod, FCH * HD, HD, 1, 3, w_prod + (size_t)(i * 2 + 1) * FCH * FCH,
               x, FCH * HD, HD, 1, N * 3, FCH);
    if (i == 1) k_add<<<(N * FCH * HD + 255) / 256, 256, 0, stream>>>(x, sc, N * FCH * HD);
    if (i == 0) k_readout0<<<N, 64, 0, stream>>>(x, w_ro0, out);
    else        k_readout1<<<N, 64, 0, stream>>>(x, w_ro1a, w_ro1b, out);
  }
}
